// ResNet_84645215470268
// MI455X (gfx1250) — matrix operand order repaired and checked by reading, not compiled, not run on hardware
//
#include <hip/hip_runtime.h>

// ---------------------------------------------------------------------------
// Sparse residual conv block for gfx1250 (MI455X), f16 WMMA w/ f32 accumulate.
//   out = conv1(relu(conv0(x))) + x,  conv(x)[i] = b + sum_k mask[i,k] x[nbr[i,k]] @ W[k]
// N=300000, C=64, K=27.  One wave32 owns a 32-row x 64-col output tile
// (two 16x64 M-tiles sharing every WMMA B fragment; EXEC all-ones throughout).
// ---------------------------------------------------------------------------

typedef __attribute__((ext_vector_type(4)))  _Float16 v4h;
typedef __attribute__((ext_vector_type(8)))  _Float16 v8h;
typedef __attribute__((ext_vector_type(16))) _Float16 v16h;
typedef __attribute__((ext_vector_type(8)))  float    v8f;
typedef __attribute__((ext_vector_type(4)))  int      v4i;

#define KOFF  27   // kernel offsets
#define CCH   64   // channels
#define CHUNK 4    // k-blocks staged per barrier round (4 * 8KB = 32KB LDS)
#define TPB   192  // 6 waves

#if __has_builtin(__builtin_amdgcn_global_load_async_to_lds_b128)
#define HAVE_ASYNC_COPY 1
typedef __attribute__((address_space(1))) v4i* as1_v4i;   // global int4*
typedef __attribute__((address_space(3))) v4i* as3_v4i;   // LDS int4*
__device__ inline void async_wait0() {
#if __has_builtin(__builtin_amdgcn_s_wait_asynccnt)
  __builtin_amdgcn_s_wait_asynccnt(0);
#else
  asm volatile("s_wait_asynccnt 0x0" ::: "memory");
#endif
}
#endif

#define SHUF16(lo, hi)                                                        \
  __builtin_shufflevector(lo, hi, 0, 1, 2, 3, 4, 5, 6, 7, 8, 9, 10, 11, 12,  \
                          13, 14, 15)

// -------------------- prep: fp32 features -> f16 ---------------------------
__global__ __launch_bounds__(256) void cvt_x_kernel(const float* __restrict__ x,
                                                    _Float16* __restrict__ xh,
                                                    int n_elems) {
  int i = (blockIdx.x * 256 + threadIdx.x) * 4;
  if (i + 3 < n_elems) {
    float4 f = *(const float4*)(x + i);
    v4h h; h[0] = (_Float16)f.x; h[1] = (_Float16)f.y;
           h[2] = (_Float16)f.z; h[3] = (_Float16)f.w;
    *(v4h*)(xh + i) = h;
  }
}

// ------------- prep: fuse validity mask into neighbor index ----------------
__global__ __launch_bounds__(256) void fuse_nbr_kernel(const int* __restrict__ nbr,
                                                       const unsigned char* __restrict__ msk,
                                                       int* __restrict__ fnbr,
                                                       int total) {
  int i = blockIdx.x * 256 + threadIdx.x;
  if (i < total) fnbr[i] = msk[i] ? nbr[i] : -1;
}

// ------------- prep: pack W[27][64][64] into WMMA-B lane layout -------------
// Packed 16B units, per k-block of 512 units:
//   unit = ((kc*4 + ntile)*2 + h)*32 + lane ; f16 slot e: j = h*8 + e
//   c_out = ntile*16 + (lane & 15)
//   c_in  = kc*32 + h*16 + ((lane >= 16) ? 8 : 0) + e     (B is K x N, column-per-lane)
__global__ __launch_bounds__(256) void pack_w_kernel(const float* __restrict__ W0,
                                                     const float* __restrict__ W1,
                                                     _Float16* __restrict__ W0p,
                                                     _Float16* __restrict__ W1p) {
  int tid = blockIdx.x * 256 + threadIdx.x;       // f16 index, 27*4096 total
  if (tid >= KOFF * CCH * CCH) return;
  int e    =  tid        & 7;
  int lane = (tid >> 3)  & 31;
  int h    = (tid >> 8)  & 1;
  int nt   = (tid >> 9)  & 3;
  int kc   = (tid >> 11) & 1;
  int k    =  tid >> 12;
  int j     = h * 8 + e;
  int c_out = nt * 16 + (lane & 15);
  int c_in  = kc * 32 + h * 16 + ((lane >> 4) << 3) + e;
  int src   = k * (CCH * CCH) + c_in * CCH + c_out;
  W0p[tid] = (_Float16)W0[src];
  W1p[tid] = (_Float16)W1[src];
}

// --------------------------- sparse conv core -------------------------------
// FINAL=false: out_h = relu(acc) as f16 (intermediate).
// FINAL=true : out_f = acc + resid (fp32).
template <bool FINAL>
__global__ __launch_bounds__(TPB) void spconv_kernel(
    const _Float16* __restrict__ xin,        // [N,64] f16 gather source
    const int* __restrict__ fnbr,            // [N,27], -1 = masked out
    const v8h* __restrict__ wpack,           // [27*512] packed B fragments
    const float* __restrict__ bias,          // [64]
    const float* __restrict__ resid,         // [N,64] fp32 (FINAL only)
    float* __restrict__ out_f,               // [N,64] fp32 (FINAL only)
    _Float16* __restrict__ out_h,            // [N,64] f16  (!FINAL only)
    int N) {
  __shared__ v8h wlds[CHUNK * 512];          // 32KB staged weights

  const int tid  = threadIdx.x;
  const int lane = tid & 31;
  const int wave = tid >> 5;
  const int rbase = (blockIdx.x * (TPB / 32) + wave) * 32;
  const bool act = rbase < N;                // wave-uniform work guard
  const int m   = lane & 15;                 // row-within-tile this lane gathers
  const int hi8 = (lane >> 4) * 8;           // lane-half select (A layout)

  // Tail rows are clamped: computed with row N-1's data, never stored.
  const int r0 = (rbase + m      < N) ? (rbase + m)      : (N - 1);
  const int r1 = (rbase + 16 + m < N) ? (rbase + 16 + m) : (N - 1);

  // C/D init with bias broadcast along rows (col = nt*16 + m)
  v8f acc[2][4];
#pragma unroll
  for (int nt = 0; nt < 4; ++nt) {
    float bv = bias[nt * 16 + m];
#pragma unroll
    for (int v = 0; v < 8; ++v) { acc[0][nt][v] = bv; acc[1][nt][v] = bv; }
  }

  for (int kb = 0; kb < KOFF; kb += CHUNK) {
    const int kn = (KOFF - kb) < CHUNK ? (KOFF - kb) : CHUNK;
    const int units = kn * 512;
    __syncthreads();                               // previous chunk's reads done
#if HAVE_ASYNC_COPY
    for (int u = tid; u < units; u += TPB)
      __builtin_amdgcn_global_load_async_to_lds_b128(
          (as1_v4i)(unsigned long long)(wpack + kb * 512 + u),
          (as3_v4i)(unsigned int)(unsigned long long)(&wlds[u]), 0, 0);
    async_wait0();
#else
    for (int u = tid; u < units; u += TPB)
      wlds[u] = wpack[kb * 512 + u];
#endif
    __syncthreads();
    if (kb + CHUNK < KOFF)                         // pull next chunk toward L2
      __builtin_prefetch((const char*)(wpack + (kb + CHUNK) * 512) + tid * 64,
                         0, 1);

    for (int kk = 0; kk < kn; ++kk) {
      const int k = kb + kk;
      if (act) {
        // ---- gather A rows for both M-tiles (fused mask: idx<0 => zero) ----
        const int f0 = fnbr[r0 * KOFF + k];
        v8h a0c[4] = {{}, {}, {}, {}};
        if (f0 >= 0) {
          const _Float16* xr = xin + (size_t)f0 * CCH;
          a0c[0] = *(const v8h*)(xr + hi8);
          a0c[1] = *(const v8h*)(xr + 16 + hi8);
          a0c[2] = *(const v8h*)(xr + 32 + hi8);
          a0c[3] = *(const v8h*)(xr + 48 + hi8);
        }
        const int f1 = fnbr[r1 * KOFF + k];
        v8h a1c[4] = {{}, {}, {}, {}};
        if (f1 >= 0) {
          const _Float16* xr = xin + (size_t)f1 * CCH;
          a1c[0] = *(const v8h*)(xr + hi8);
          a1c[1] = *(const v8h*)(xr + 16 + hi8);
          a1c[2] = *(const v8h*)(xr + 32 + hi8);
          a1c[3] = *(const v8h*)(xr + 48 + hi8);
        }
        // ---- 2 K-chunks x 4 N-tiles, B shared by both M-tiles ----
#pragma unroll
        for (int kc = 0; kc < 2; ++kc) {
          v16h A0 = SHUF16(a0c[2 * kc], a0c[2 * kc + 1]);
          v16h A1 = SHUF16(a1c[2 * kc], a1c[2 * kc + 1]);
          v16h Bv[4];
#pragma unroll
          for (int nt = 0; nt < 4; ++nt) {         // all B loads first
            int bbase = kk * 512 + ((kc * 4 + nt) * 2) * 32 + lane;
            Bv[nt] = SHUF16(wlds[bbase], wlds[bbase + 32]);
          }
#pragma unroll
          for (int nt = 0; nt < 4; ++nt) {         // 8 back-to-back WMMAs
            acc[0][nt] = __builtin_amdgcn_wmma_f32_16x16x32_f16(
                false, A0, false, Bv[nt], (short)0, acc[0][nt], false, false);
            acc[1][nt] = __builtin_amdgcn_wmma_f32_16x16x32_f16(
                false, A1, false, Bv[nt], (short)0, acc[1][nt], false, false);
          }
        }
      }
    }
  }

  // C/D layout: VGPR v -> row v (lanes 0-15) / row v+8 (lanes 16-31)
#pragma unroll
  for (int t = 0; t < 2; ++t) {
    if (rbase + t * 16 < N) {                      // wave-uniform store guard
#pragma unroll
      for (int nt = 0; nt < 4; ++nt) {
        int col = nt * 16 + m;
#pragma unroll
        for (int v = 0; v < 8; ++v) {
          size_t off = (size_t)(rbase + t * 16 + v + hi8) * CCH + col;
          float val = acc[t][nt][v];
          if (FINAL) out_f[off] = val + resid[off];
          else       out_h[off] = (_Float16)fmaxf(val, 0.0f);
        }
      }
    }
  }
}

// ------------------------------- launcher -----------------------------------
extern "C" void kernel_launch(void* const* d_in, const int* in_sizes, int n_in,
                              void* d_out, int out_size, void* d_ws, size_t ws_size,
                              hipStream_t stream) {
  const float*         feat = (const float*)d_in[0];
  const int*           nbr  = (const int*)d_in[1];
  const unsigned char* mk   = (const unsigned char*)d_in[2];
  const float*         W0   = (const float*)d_in[3];
  const float*         b0   = (const float*)d_in[4];
  const float*         W1   = (const float*)d_in[5];
  const float*         b1   = (const float*)d_in[6];
  float*               out  = (float*)d_out;

  const int N = in_sizes[0] / CCH;
  const size_t WBYTES = (size_t)KOFF * CCH * CCH * 2;   // packed W, f16

  // Workspace carve-up: xh [N,64]f16 | yh [N,64]f16 | W0p | W1p | fnbr [N,27]i32
  char* ws = (char*)d_ws;
  _Float16* xh   = (_Float16*)ws;
  _Float16* yh   = (_Float16*)(ws + (size_t)N * CCH * 2);
  _Float16* w0p  = (_Float16*)(ws + (size_t)N * CCH * 4);
  _Float16* w1p  = (_Float16*)(ws + (size_t)N * CCH * 4 + WBYTES);
  int*      fnbr = (int*)     (ws + (size_t)N * CCH * 4 + 2 * WBYTES);

  // prep
  int ncv = N * CCH;
  cvt_x_kernel<<<(ncv / 4 + 255) / 256, 256, 0, stream>>>(feat, xh, ncv);
  int nnk = N * KOFF;
  fuse_nbr_kernel<<<(nnk + 255) / 256, 256, 0, stream>>>(nbr, mk, fnbr, nnk);
  pack_w_kernel<<<(KOFF * CCH * CCH + 255) / 256, 256, 0, stream>>>(W0, W1, w0p, w1p);

  // conv0 + relu -> yh (f16), then conv1 + bias + residual -> out (fp32)
  int nwaves = (N + 31) / 32;
  int blocks = (nwaves + (TPB / 32) - 1) / (TPB / 32);
  spconv_kernel<false><<<blocks, TPB, 0, stream>>>(
      xh, fnbr, (const v8h*)w0p, b0, nullptr, nullptr, yh, N);
  spconv_kernel<true><<<blocks, TPB, 0, stream>>>(
      yh, fnbr, (const v8h*)w1p, b1, feat, out, nullptr, N);
}
